// MultiHeadAttention_41034117546200
// MI455X (gfx1250) — matrix operand order repaired and checked by reading, not compiled, not run on hardware
//
#include <hip/hip_runtime.h>
#include <hip/hip_bf16.h>

// ---------------------------------------------------------------------------
// Types for CDNA5 WMMA (wave32): v_wmma_f32_16x16x32_bf16
// ---------------------------------------------------------------------------
typedef __bf16 bf16_t;
typedef __attribute__((ext_vector_type(16))) __bf16 v16bf;
typedef __attribute__((ext_vector_type(8)))  __bf16 v8bf;
typedef __attribute__((ext_vector_type(8)))  float  v8f;

#define GB 2
#define GT 2048
#define GE 1024
#define GH 16
#define GD 64
#define GF (GH * GD)          // 1024
#define GM (GB * GT)          // 4096
#define BIAS_W 4096           // bias table width per head (rel + 2048)

__device__ __forceinline__ v8f v8f_zero() {
    v8f z = {0.f, 0.f, 0.f, 0.f, 0.f, 0.f, 0.f, 0.f};
    return z;
}

__device__ __forceinline__ v16bf combine16(v8bf lo, v8bf hi) {
    v16bf r;
#pragma unroll
    for (int i = 0; i < 8; ++i) { r[i] = lo[i]; r[i + 8] = hi[i]; }
    return r;
}

// A-matrix fragment, 16x32 bf16 (M x K), base = &A[row0*ld + k0].
// ISA layout: lanes 0-15 (M=lane): K 0-7 in v0-3, K 16-23 in v4-7;
//             lanes 16-31 (M=lane-16): K 8-15 and K 24-31.
__device__ __forceinline__ v16bf load_afrag(const bf16_t* __restrict__ base,
                                            int ld, int lane) {
    const int m  = lane & 15;
    const int kb = (lane >> 4) << 3;           // 0 or 8
    const bf16_t* p = base + (size_t)m * ld + kb;
    v8bf lo = *(const v8bf*)(p);               // K = kb .. kb+7
    v8bf hi = *(const v8bf*)(p + 16);          // K = kb+16 .. kb+23
    return combine16(lo, hi);
}

// B-matrix fragment, 32x16 bf16 (K x N), sourced from B^T stored [N,K]
// row-major; base = &Bt[n_row0*ld + k0].
// ISA layout: lanes 0-15 (N=lane): K 0-7 and K 16-23; lanes 16-31: K 8-15 and K 24-31.
__device__ __forceinline__ v16bf load_bfrag(const bf16_t* __restrict__ base,
                                            int ld, int lane) {
    const int n  = lane & 15;
    const int kb = (lane >> 4) << 3;           // 0 or 8
    const bf16_t* p = base + (size_t)n * ld + kb;
    v8bf lo = *(const v8bf*)(p);
    v8bf hi = *(const v8bf*)(p + 16);
    return combine16(lo, hi);
}

__device__ __forceinline__ v8f wmma_bf16(v16bf a, v16bf b, v8f c) {
    return __builtin_amdgcn_wmma_f32_16x16x32_bf16(
        /*neg_a=*/false, a, /*neg_b=*/false, b,
        /*c_mod=*/(short)0, c, /*reuse_a=*/false, /*reuse_b=*/false);
}

// ---------------------------------------------------------------------------
// fp32 -> bf16 conversion
// ---------------------------------------------------------------------------
__global__ void cvt_f32_bf16(const float* __restrict__ in,
                             bf16_t* __restrict__ out, int n) {
    int i = blockIdx.x * blockDim.x + threadIdx.x;
    if (i < n) out[i] = (bf16_t)in[i];
}

// ---------------------------------------------------------------------------
// T5 relative-position bias table: tab[h][rel + 2048] = rel_emb[bucket(rel), h]
// ---------------------------------------------------------------------------
__global__ void bias_table_kernel(const float* __restrict__ rel_emb,
                                  float* __restrict__ tab) {
    int idx = blockIdx.x * blockDim.x + threadIdx.x;
    if (idx >= GH * BIAS_W) return;
    int h   = idx >> 12;
    int rel = (idx & (BIAS_W - 1)) - GT;       // t - s
    int bucket = (rel > 0) ? 16 : 0;           // nb = 16
    int a = rel < 0 ? -rel : rel;
    int v;
    if (a < 8) {                               // max_exact = 8
        v = a;
    } else {
        v = 8 + (int)(__logf((float)a * 0.125f) * (8.0f / __logf(16.0f)));
        if (v > 15) v = 15;
    }
    tab[idx] = rel_emb[(bucket + v) * GH + h];
}

// ---------------------------------------------------------------------------
// Tiled bf16 WMMA GEMM: C[m,n] = sum_k A[m,k] * Bt[n,k]
//   A:  [M,K] bf16 row-major, Bt: [N,K] bf16 row-major.
//   Block = 256 threads (8 waves): 128x128 tile; wave = 32x64 (2x4 WMMA tiles).
//   mode 0: Cf[m*N+n] = f32
//   mode 1: Cb[((b*H+h)*T+t)*D+d]    (q/k layout, b=m/T, t=m%T, h=n/D, d=n%D)
//   mode 2: Cb[((b*H+h)*D+d)*T+t]    (V^T layout)
// ---------------------------------------------------------------------------
__global__ __launch_bounds__(256)
void gemm_bf16_kernel(const bf16_t* __restrict__ A, const bf16_t* __restrict__ Bt,
                      float* __restrict__ Cf, bf16_t* __restrict__ Cb,
                      int M, int N, int K, int mode) {
    const int lane = threadIdx.x & 31;
    const int wave = threadIdx.x >> 5;
    const int wm = wave & 3;               // 4 waves along M
    const int wn = wave >> 2;              // 2 waves along N
    const int m0 = blockIdx.y * 128 + wm * 32;
    const int n0 = blockIdx.x * 128 + wn * 64;

    v8f acc[2][4];
#pragma unroll
    for (int mt = 0; mt < 2; ++mt)
#pragma unroll
        for (int nt = 0; nt < 4; ++nt) acc[mt][nt] = v8f_zero();

    for (int k0 = 0; k0 < K; k0 += 32) {
        v16bf af0 = load_afrag(A + (size_t)m0 * K + k0, K, lane);
        v16bf af1 = load_afrag(A + (size_t)(m0 + 16) * K + k0, K, lane);
#pragma unroll
        for (int nt = 0; nt < 4; ++nt) {
            v16bf bfm = load_bfrag(Bt + (size_t)(n0 + nt * 16) * K + k0, K, lane);
            acc[0][nt] = wmma_bf16(af0, bfm, acc[0][nt]);
            acc[1][nt] = wmma_bf16(af1, bfm, acc[1][nt]);
        }
    }

    const int nn  = lane & 15;
    const int hi8 = (lane >> 4) << 3;
#pragma unroll
    for (int mt = 0; mt < 2; ++mt)
#pragma unroll
        for (int nt = 0; nt < 4; ++nt)
#pragma unroll
            for (int r = 0; r < 8; ++r) {
                int m = m0 + mt * 16 + hi8 + r;
                int n = n0 + nt * 16 + nn;
                float val = acc[mt][nt][r];
                if (mode == 0) {
                    Cf[(size_t)m * N + n] = val;
                } else {
                    int b = m >> 11, t = m & (GT - 1);
                    int h = n >> 6,  d = n & (GD - 1);
                    size_t addr = (mode == 1)
                        ? ((((size_t)b * GH + h) * GT + t) * GD + d)
                        : ((((size_t)b * GH + h) * GD + d) * GT + t);
                    Cb[addr] = (bf16_t)val;
                }
            }
}

// ---------------------------------------------------------------------------
// Flash attention with T5 bias + causal mask.
//   q,k: [B,H,T,D] bf16.  vT: [B,H,D,T] bf16.  out: [B,T,H*D] bf16.
//   Block = 256 threads = 8 waves; block covers 128 query rows; each wave
//   owns 16 rows. Key loop in blocks of 64 (4 WMMA N-tiles).
// ---------------------------------------------------------------------------
__global__ __launch_bounds__(256)
void attn_kernel(const bf16_t* __restrict__ q, const bf16_t* __restrict__ k,
                 const bf16_t* __restrict__ vT, const float* __restrict__ bias_tab,
                 bf16_t* __restrict__ out) {
    __shared__ bf16_t pbuf[8 * 16 * 64];       // per-wave P staging (16 KB)

    const int lane = threadIdx.x & 31;
    const int wave = threadIdx.x >> 5;
    const int nqb  = GT / 128;                 // 16 query blocks per (b,h)
    const int qblk = blockIdx.x % nqb;
    const int bh   = blockIdx.x / nqb;
    const int h    = bh % GH;
    const int b    = bh / GH;

    const bf16_t* qp = q  + (size_t)bh * GT * GD;
    const bf16_t* kp = k  + (size_t)bh * GT * GD;
    const bf16_t* vp = vT + (size_t)bh * GD * GT;
    const float*  bt = bias_tab + (size_t)h * BIAS_W;

    const int t0  = qblk * 128 + wave * 16;
    const int nn  = lane & 15;
    const int hi8 = (lane >> 4) << 3;

    // q fragments are invariant over the key loop (K = D = 64 -> 2 fragments)
    v16bf qf0 = load_afrag(qp + (size_t)t0 * GD + 0,  GD, lane);
    v16bf qf1 = load_afrag(qp + (size_t)t0 * GD + 32, GD, lane);

    v8f oacc[4];
#pragma unroll
    for (int dt = 0; dt < 4; ++dt) oacc[dt] = v8f_zero();

    float rmax[8], rsum[8];
#pragma unroll
    for (int r = 0; r < 8; ++r) { rmax[r] = -1e30f; rsum[r] = 0.0f; }

    bf16_t* pl = &pbuf[wave * 16 * 64];
    const int s_end = qblk * 128 + 128;        // causal: keys beyond block's max t unused

    for (int s0 = 0; s0 < s_end; s0 += 64) {
        // ---- S = q * k^T  (4 N-tiles of 16 keys, K over D in 2 steps) ----
        v8f sf[4];
#pragma unroll
        for (int nt = 0; nt < 4; ++nt) sf[nt] = v8f_zero();
#pragma unroll
        for (int nt = 0; nt < 4; ++nt) {
            v16bf kf0 = load_bfrag(kp + (size_t)(s0 + nt * 16) * GD + 0,  GD, lane);
            v16bf kf1 = load_bfrag(kp + (size_t)(s0 + nt * 16) * GD + 32, GD, lane);
            sf[nt] = wmma_bf16(qf0, kf0, sf[nt]);
            sf[nt] = wmma_bf16(qf1, kf1, sf[nt]);
        }

        // ---- scale + bias + causal mask, block row max ----
        float bmax[8];
#pragma unroll
        for (int r = 0; r < 8; ++r) bmax[r] = -1e30f;
#pragma unroll
        for (int nt = 0; nt < 4; ++nt)
#pragma unroll
            for (int r = 0; r < 8; ++r) {
                int t = t0 + hi8 + r;
                int s = s0 + nt * 16 + nn;
                float val = sf[nt][r] * 0.125f + bt[t - s + GT];
                if (s > t) val = -1e30f;
                sf[nt][r] = val;
                bmax[r] = fmaxf(bmax[r], val);
            }
        // rows live across 16 lanes (same half-wave): xor-reduce within group
#pragma unroll
        for (int r = 0; r < 8; ++r) {
            bmax[r] = fmaxf(bmax[r], __shfl_xor(bmax[r], 8, 32));
            bmax[r] = fmaxf(bmax[r], __shfl_xor(bmax[r], 4, 32));
            bmax[r] = fmaxf(bmax[r], __shfl_xor(bmax[r], 2, 32));
            bmax[r] = fmaxf(bmax[r], __shfl_xor(bmax[r], 1, 32));
        }

        // ---- online softmax update ----
        float corr[8];
#pragma unroll
        for (int r = 0; r < 8; ++r) {
            float nm = fmaxf(rmax[r], bmax[r]);
            corr[r] = __expf(rmax[r] - nm);
            rmax[r] = nm;
        }
        float bsum[8];
#pragma unroll
        for (int r = 0; r < 8; ++r) bsum[r] = 0.0f;
#pragma unroll
        for (int nt = 0; nt < 4; ++nt)
#pragma unroll
            for (int r = 0; r < 8; ++r) {
                float p = __expf(sf[nt][r] - rmax[r]);
                sf[nt][r] = p;
                bsum[r] += p;
            }
#pragma unroll
        for (int r = 0; r < 8; ++r) {
            bsum[r] += __shfl_xor(bsum[r], 8, 32);
            bsum[r] += __shfl_xor(bsum[r], 4, 32);
            bsum[r] += __shfl_xor(bsum[r], 2, 32);
            bsum[r] += __shfl_xor(bsum[r], 1, 32);
            rsum[r] = rsum[r] * corr[r] + bsum[r];
        }
#pragma unroll
        for (int dt = 0; dt < 4; ++dt)
#pragma unroll
            for (int r = 0; r < 8; ++r) oacc[dt][r] *= corr[r];

        // ---- repack P (C-layout -> A-layout) through per-wave LDS ----
#pragma unroll
        for (int nt = 0; nt < 4; ++nt)
#pragma unroll
            for (int r = 0; r < 8; ++r)
                pl[(hi8 + r) * 64 + nt * 16 + nn] = (bf16_t)sf[nt][r];
        // wave-synchronous LDS: lanes in one wave share the staging tile
        asm volatile("s_wait_dscnt 0x0" ::: "memory");

        // ---- O += P * V  (B operand from V^T [D,T] layout) ----
#pragma unroll
        for (int kk = 0; kk < 2; ++kk) {
            v16bf pa = load_afrag(pl + kk * 32, 64, lane);
#pragma unroll
            for (int dt = 0; dt < 4; ++dt) {
                v16bf vf = load_bfrag(vp + (size_t)(dt * 16) * GT + s0 + kk * 32,
                                      GT, lane);
                oacc[dt] = wmma_bf16(pa, vf, oacc[dt]);
            }
        }

        // prefetch next key block (global_prefetch_b8)
        if (s0 + 64 < s_end)
            __builtin_prefetch(kp + (size_t)(s0 + 64) * GD + nn * 128, 0, 3);
    }

    // ---- normalize and store O in [B, T, H*D] bf16 ----
#pragma unroll
    for (int r = 0; r < 8; ++r) rsum[r] = 1.0f / rsum[r];
#pragma unroll
    for (int dt = 0; dt < 4; ++dt)
#pragma unroll
        for (int r = 0; r < 8; ++r) {
            int t = t0 + hi8 + r;
            int d = dt * 16 + nn;
            out[(((size_t)b * GT + t) * GH + h) * GD + d] =
                (bf16_t)(oacc[dt][r] * rsum[r]);
        }
}

// ---------------------------------------------------------------------------
// Host-side orchestration
// ---------------------------------------------------------------------------
extern "C" void kernel_launch(void* const* d_in, const int* in_sizes, int n_in,
                              void* d_out, int out_size, void* d_ws, size_t ws_size,
                              hipStream_t stream) {
    const float* x       = (const float*)d_in[0];   // [B,T,E]
    const float* Wq      = (const float*)d_in[1];   // [F,E]
    const float* Wk      = (const float*)d_in[2];
    const float* Wv      = (const float*)d_in[3];
    const float* Wo      = (const float*)d_in[4];   // [E,F]
    const float* rel_emb = (const float*)d_in[5];   // [32,H]
    float* outp = (float*)d_out;                    // [B,T,E] f32

    char* ws = (char*)d_ws;
    const size_t SZ_X = (size_t)GM * GE * 2;        // 8 MiB bf16
    const size_t SZ_W = (size_t)GE * GF * 2;        // 2 MiB bf16
    bf16_t* xb   = (bf16_t*)(ws);
    bf16_t* wqb  = (bf16_t*)(ws + SZ_X);
    bf16_t* wkb  = (bf16_t*)(ws + SZ_X + 1 * SZ_W);
    bf16_t* wvb  = (bf16_t*)(ws + SZ_X + 2 * SZ_W);
    bf16_t* wob  = (bf16_t*)(ws + SZ_X + 3 * SZ_W);
    bf16_t* qb   = (bf16_t*)(ws + SZ_X + 4 * SZ_W);
    bf16_t* kb   = (bf16_t*)(ws + 2 * SZ_X + 4 * SZ_W);
    bf16_t* vtb  = (bf16_t*)(ws + 3 * SZ_X + 4 * SZ_W);
    bf16_t* ob   = (bf16_t*)(ws + 4 * SZ_X + 4 * SZ_W);
    float*  btab = (float*) (ws + 5 * SZ_X + 4 * SZ_W);

    // 1) fp32 -> bf16
    {
        int nx = GM * GE;
        cvt_f32_bf16<<<(nx + 255) / 256, 256, 0, stream>>>(x, xb, nx);
        int nw = GE * GF;
        cvt_f32_bf16<<<(nw + 255) / 256, 256, 0, stream>>>(Wq, wqb, nw);
        cvt_f32_bf16<<<(nw + 255) / 256, 256, 0, stream>>>(Wk, wkb, nw);
        cvt_f32_bf16<<<(nw + 255) / 256, 256, 0, stream>>>(Wv, wvb, nw);
        cvt_f32_bf16<<<(nw + 255) / 256, 256, 0, stream>>>(Wo, wob, nw);
    }

    // 2) T5 bias table [H, 4096]
    bias_table_kernel<<<(GH * BIAS_W) / 256, 256, 0, stream>>>(rel_emb, btab);

    // 3) QKV projections (WMMA GEMMs)
    dim3 gproj(GF / 128, GM / 128);                 // (8, 32)
    gemm_bf16_kernel<<<gproj, 256, 0, stream>>>(xb, wqb, nullptr, qb,
                                                GM, GF, GE, 1);
    gemm_bf16_kernel<<<gproj, 256, 0, stream>>>(xb, wkb, nullptr, kb,
                                                GM, GF, GE, 1);
    gemm_bf16_kernel<<<gproj, 256, 0, stream>>>(xb, wvb, nullptr, vtb,
                                                GM, GF, GE, 2);

    // 4) flash attention: B*H*(T/128) = 512 workgroups
    attn_kernel<<<GB * GH * (GT / 128), 256, 0, stream>>>(qb, kb, vtb, btab, ob);

    // 5) output projection -> f32 d_out
    dim3 gout(GE / 128, GM / 128);
    gemm_bf16_kernel<<<gout, 256, 0, stream>>>(ob, wob, outp, nullptr,
                                               GM, GE, GF, 0);
}
